// LeapfrogIntegrator_81466939671260
// MI455X (gfx1250) — compile-verified, matrix operand order checked by reading, not run on hardware
//
#include <hip/hip_runtime.h>
#include <hip/hip_bf16.h>

// ---------------------------------------------------------------------------
// Leapfrog integrator: low-rank Christoffel force + learned friction.
// B=4096, D=1024, R=256, steps=8, dt=0.01, h=0.005.
// bf16 GEMMs via v_wmma_f32_16x16x32_bf16, f32 state lives in d_out.
// Per-wave tile: 32 rows x 64 cols (8 accumulators). K loop unrolled x2 with
// ping-pong fragment sets: no register copies, loads stay in flight behind
// the matrix pipe (partial s_wait_loadcnt).
// ---------------------------------------------------------------------------

typedef __attribute__((ext_vector_type(16))) __bf16 v16bf;
typedef __attribute__((ext_vector_type(8)))  float  v8f;

#define BROWS 4096
#define DDIM  1024
#define RDIM  256
#define DT_F  0.01f
#define H_F   0.005f

__device__ __forceinline__ unsigned short f32_to_bf16u(float f) {
  unsigned int u = __float_as_uint(f);
  u += 0x7FFFu + ((u >> 16) & 1u);   // round-to-nearest-even
  return (unsigned short)(u >> 16);
}

union FragU { v16bf bf; uint4 q[2]; };

// One 16x32 bf16 fragment row for this lane (ISA 7.12.2 16-bit layout):
// lane%16 = row, lane/16 = khalf; VGPR0-3: K=khalf*8+0..7, VGPR4-7: +16.
// Caller pre-bakes (row*ld + khalf*8) into the pointer.
__device__ __forceinline__ void load2(v16bf& dst, const unsigned short* p) {
  FragU f;
  f.q[0] = *reinterpret_cast<const uint4*>(p);
  f.q[1] = *reinterpret_cast<const uint4*>(p + 16);
  dst = f.bf;
}

struct FragSet { v16bf a0, a1, b0, b1, b2, b3; };

__device__ __forceinline__ void load_set(FragSet& s,
    const unsigned short* ap0, const unsigned short* ap1,
    const unsigned short* bp, size_t bstride, int k0) {
  load2(s.a0, ap0 + k0);
  load2(s.a1, ap1 + k0);
  load2(s.b0, bp + k0);
  load2(s.b1, bp + bstride + k0);
  load2(s.b2, bp + 2 * bstride + k0);
  load2(s.b3, bp + 3 * bstride + k0);
}

__device__ __forceinline__ void mma_set(v8f acc[2][4], const FragSet& s) {
  acc[0][0] = __builtin_amdgcn_wmma_f32_16x16x32_bf16(false, s.a0, false, s.b0, (short)0, acc[0][0], false, false);
  acc[0][1] = __builtin_amdgcn_wmma_f32_16x16x32_bf16(false, s.a0, false, s.b1, (short)0, acc[0][1], false, false);
  acc[0][2] = __builtin_amdgcn_wmma_f32_16x16x32_bf16(false, s.a0, false, s.b2, (short)0, acc[0][2], false, false);
  acc[0][3] = __builtin_amdgcn_wmma_f32_16x16x32_bf16(false, s.a0, false, s.b3, (short)0, acc[0][3], false, false);
  acc[1][0] = __builtin_amdgcn_wmma_f32_16x16x32_bf16(false, s.a1, false, s.b0, (short)0, acc[1][0], false, false);
  acc[1][1] = __builtin_amdgcn_wmma_f32_16x16x32_bf16(false, s.a1, false, s.b1, (short)0, acc[1][1], false, false);
  acc[1][2] = __builtin_amdgcn_wmma_f32_16x16x32_bf16(false, s.a1, false, s.b2, (short)0, acc[1][2], false, false);
  acc[1][3] = __builtin_amdgcn_wmma_f32_16x16x32_bf16(false, s.a1, false, s.b3, (short)0, acc[1][3], false, false);
}

// Ping-pong pipelined K loop (kdim/32 must be even, true for 1024 and 256):
// f0/f1 alternate between "being loaded" and "being consumed" -> no copies.
__device__ __forceinline__ void gemm_loop(v8f acc[2][4],
    const unsigned short* ap0, const unsigned short* ap1,
    const unsigned short* bp, size_t bstride, int kdim) {
  const int N = kdim >> 5;   // number of 32-wide K steps (even)
  FragSet f0, f1;
  load_set(f0, ap0, ap1, bp, bstride, 0);
  for (int s = 0; s + 2 < N; s += 2) {
    load_set(f1, ap0, ap1, bp, bstride, (s + 1) << 5);
    mma_set(acc, f0);
    load_set(f0, ap0, ap1, bp, bstride, (s + 2) << 5);
    mma_set(acc, f1);
  }
  load_set(f1, ap0, ap1, bp, bstride, (N - 1) << 5);
  mma_set(acc, f0);
  mma_set(acc, f1);
}

// ---------------------------------------------------------------------------
// K1: T[B,R] = tanh( X[B,D] @ U[R,D]^T ).
// Block = 8 waves; wave owns 32 rows x 64 cols. Grid: (R/64, B/256).
// ---------------------------------------------------------------------------
__global__ __launch_bounds__(256) void k_tanh_xu(
    const unsigned short* __restrict__ xbf,
    const unsigned short* __restrict__ ubf,
    unsigned short* __restrict__ tout) {
  const int lane  = threadIdx.x & 31;
  const int wave  = threadIdx.x >> 5;
  const int khalf = lane >> 4;
  const int l16   = lane & 15;
  const int row0  = blockIdx.y * 256 + wave * 32;
  const int col0  = blockIdx.x * 64;

  const unsigned short* ap0 = xbf + (size_t)(row0 + l16) * DDIM + khalf * 8;
  const unsigned short* ap1 = ap0 + (size_t)16 * DDIM;
  const unsigned short* bp  = ubf + (size_t)(col0 + l16) * DDIM + khalf * 8;

  v8f acc[2][4] = {};
  gemm_loop(acc, ap0, ap1, bp, (size_t)16 * DDIM, DDIM);

#pragma unroll
  for (int r = 0; r < 2; ++r)
#pragma unroll
    for (int i = 0; i < 4; ++i) {
      const int n = col0 + 16 * i + l16;
#pragma unroll
      for (int j = 0; j < 8; ++j) {
        const int m = row0 + 16 * r + khalf * 8 + j;
        tout[(size_t)m * RDIM + n] = f32_to_bf16u(tanhf(acc[r][i][j]));
      }
    }
}

// ---------------------------------------------------------------------------
// K2: per 32x64 tile, fused:
//   mu_mat = X @ Wf^T (K=1024);  g_mat = T @ W^T (K=256)
//   mu = 5*sigmoid(mu_mat + bf[n]); gamma = g_mat * v
//   v_new = (v + h*(force - gamma)) / (1 + h*mu)            (in place)
//   if update_x: x_new = x + dt*v_new -> xs (in place), xbf_out (ping-pong)
// ---------------------------------------------------------------------------
__global__ __launch_bounds__(256) void k_fused_update(
    const unsigned short* __restrict__ xbf,
    const unsigned short* __restrict__ tmat,
    const unsigned short* __restrict__ wfbf,
    const unsigned short* __restrict__ wbf,
    const float* __restrict__ bfv,
    const float* __restrict__ force,
    float* __restrict__ xs,
    float* __restrict__ vs,
    unsigned short* __restrict__ xbf_out,
    int update_x) {
  const int lane  = threadIdx.x & 31;
  const int wave  = threadIdx.x >> 5;
  const int khalf = lane >> 4;
  const int l16   = lane & 15;
  const int row0  = blockIdx.y * 256 + wave * 32;
  const int col0  = blockIdx.x * 64;

  const unsigned short* ap0 = xbf + (size_t)(row0 + l16) * DDIM + khalf * 8;
  const unsigned short* ap1 = ap0 + (size_t)16 * DDIM;
  const unsigned short* bp  = wfbf + (size_t)(col0 + l16) * DDIM + khalf * 8;

  v8f macc[2][4] = {};
  gemm_loop(macc, ap0, ap1, bp, (size_t)16 * DDIM, DDIM);

  const unsigned short* tp0 = tmat + (size_t)(row0 + l16) * RDIM + khalf * 8;
  const unsigned short* tp1 = tp0 + (size_t)16 * RDIM;
  const unsigned short* wp  = wbf + (size_t)(col0 + l16) * RDIM + khalf * 8;

  v8f gacc[2][4] = {};
  gemm_loop(gacc, tp0, tp1, wp, (size_t)16 * RDIM, RDIM);

#pragma unroll
  for (int r = 0; r < 2; ++r)
#pragma unroll
    for (int i = 0; i < 4; ++i) {
      const int n = col0 + 16 * i + l16;
      const float bias = bfv[n];
#pragma unroll
      for (int j = 0; j < 8; ++j) {
        const int m = row0 + 16 * r + khalf * 8 + j;
        const size_t idx = (size_t)m * DDIM + n;
        const float mu  = 5.0f / (1.0f + __expf(-(macc[r][i][j] + bias)));
        const float vin = vs[idx];
        const float g   = gacc[r][i][j] * vin;
        const float vh  = (vin + H_F * (force[idx] - g)) / (1.0f + H_F * mu);
        vs[idx] = vh;
        if (update_x) {
          const float xn = xs[idx] + DT_F * vh;
          xs[idx] = xn;
          xbf_out[idx] = f32_to_bf16u(xn);
        }
      }
    }
}

// ---------------------------------------------------------------------------
// Helpers: f32 -> bf16 weight conversion; state initialization.
// ---------------------------------------------------------------------------
__global__ void k_cvt_bf16(const float* __restrict__ src,
                           unsigned short* __restrict__ dst, int n) {
  int i = blockIdx.x * blockDim.x + threadIdx.x;
  if (i < n) dst[i] = f32_to_bf16u(src[i]);
}

__global__ void k_init_state(const float* __restrict__ x,
                             const float* __restrict__ v,
                             float* __restrict__ xs, float* __restrict__ vs,
                             unsigned short* __restrict__ xbf, int n) {
  int i = blockIdx.x * blockDim.x + threadIdx.x;
  if (i < n) {
    float xv = x[i];
    xs[i] = xv;
    vs[i] = v[i];
    xbf[i] = f32_to_bf16u(xv);
  }
}

// ---------------------------------------------------------------------------
extern "C" void kernel_launch(void* const* d_in, const int* in_sizes, int n_in,
                              void* d_out, int out_size, void* d_ws,
                              size_t ws_size, hipStream_t stream) {
  (void)in_sizes; (void)n_in; (void)out_size; (void)ws_size;
  const float* x     = (const float*)d_in[0];
  const float* v     = (const float*)d_in[1];
  const float* force = (const float*)d_in[2];
  const float* U     = (const float*)d_in[3];
  const float* W     = (const float*)d_in[4];
  const float* Wf    = (const float*)d_in[5];
  const float* bf    = (const float*)d_in[6];
  const int steps = 8;  // fixed by setup_inputs()

  // f32 state lives directly in d_out: [x | v]
  float* xs = (float*)d_out;
  float* vs = xs + (size_t)BROWS * DDIM;

  // Workspace layout (all 2-byte, offsets naturally aligned):
  unsigned short* xbfA = (unsigned short*)d_ws;              // B*D
  unsigned short* xbfB = xbfA + (size_t)BROWS * DDIM;        // B*D
  unsigned short* T    = xbfB + (size_t)BROWS * DDIM;        // B*R
  unsigned short* Ubf  = T    + (size_t)BROWS * RDIM;        // R*D
  unsigned short* Wbf  = Ubf  + (size_t)RDIM * DDIM;         // D*R
  unsigned short* Wfbf = Wbf  + (size_t)DDIM * RDIM;         // D*D

  int n;
  n = RDIM * DDIM;  k_cvt_bf16<<<(n + 255) / 256, 256, 0, stream>>>(U, Ubf, n);
  n = DDIM * RDIM;  k_cvt_bf16<<<(n + 255) / 256, 256, 0, stream>>>(W, Wbf, n);
  n = DDIM * DDIM;  k_cvt_bf16<<<(n + 255) / 256, 256, 0, stream>>>(Wf, Wfbf, n);
  n = BROWS * DDIM;
  k_init_state<<<(n + 255) / 256, 256, 0, stream>>>(x, v, xs, vs, xbfA, n);

  const dim3 blk(256);                       // 8 waves (wave32)
  const dim3 g1(RDIM / 64, BROWS / 256);     // (4, 16)
  const dim3 g2(DDIM / 64, BROWS / 256);     // (16, 16)

  unsigned short* cur = xbfA;
  unsigned short* nxt = xbfB;
  for (int s = 0; s < steps; ++s) {
    // first half-step: uses (cx, cv); updates v -> v_half, x -> nx
    k_tanh_xu<<<g1, blk, 0, stream>>>(cur, Ubf, T);
    k_fused_update<<<g2, blk, 0, stream>>>(cur, T, Wfbf, Wbf, bf, force,
                                           xs, vs, nxt, 1);
    { unsigned short* t = cur; cur = nxt; nxt = t; }
    // second half-step: uses (nx, v_half); updates v -> nv
    k_tanh_xu<<<g1, blk, 0, stream>>>(cur, Ubf, T);
    k_fused_update<<<g2, blk, 0, stream>>>(cur, T, Wfbf, Wbf, bf, force,
                                           xs, vs, nxt, 0);
  }
}
